// BeamDecoder_10050223473155
// MI455X (gfx1250) — matrix operand order repaired and checked by reading, not compiled, not run on hardware
//
#include <hip/hip_runtime.h>
#include <hip/hip_bf16.h>
#include <math.h>

// ---------------- problem constants ----------------
#define BB 16
#define SS 64
#define VV 32000
#define DD 1024
#define WW 4
#define NROW (BB*WW)          // 64 beam rows
#define ML 75                 // max_len = min(int(1.1*64+5), 512)
#define EOS_ID 2
#define NEGF (-1.0e9f)
#define FNEG (-3.4e38f)
#define CHUNK 128             // columns per GEMM block
#define NCHUNK (VV/CHUNK)     // 250
#define KTILE 32
#define NKIT (DD/KTILE)       // 32

typedef __attribute__((ext_vector_type(16))) _Float16 v16h;
typedef __attribute__((ext_vector_type(8)))  _Float16 v8h;   // 16 bytes
typedef __attribute__((ext_vector_type(8)))  float    v8f;
typedef __attribute__((ext_vector_type(4)))  unsigned int v4u;
typedef __attribute__((ext_vector_type(8)))  int      v8i_t;
typedef __attribute__((ext_vector_type(4)))  int      v4i_t;

// async per-thread LDS staging (verified to assemble in round 1)
#define USE_ASYNC_LDS 1
// Tensor Data Mover for the B (weight) stream
#ifndef USE_TDM
#if __has_builtin(__builtin_amdgcn_tensor_load_to_lds)
#define USE_TDM 1
#else
#define USE_TDM 0
#endif
#endif
// this toolchain's builtin arity (compile error round 2 said: expected 6)
#ifndef TDM_ARITY
#define TDM_ARITY 6
#endif

__device__ __forceinline__ void wait_tensorcnt0() {
#if __has_builtin(__builtin_amdgcn_s_wait_tensorcnt)
  __builtin_amdgcn_s_wait_tensorcnt(0);
#else
  asm volatile("s_wait_tensorcnt 0x0" ::: "memory");
#endif
}

#if USE_TDM
// 2D TDM load: tile_dim1=128 rows x tile_dim0=32 half elements, row stride 1024 elems.
// D# per CDNA5 ISA 8.3/8.4: group0 {count=1 | lds_addr | global_addr | type=2},
// group1 {data_size=2B, tensor dims (huge -> no OOB), tile dims, dim0 stride}.
__device__ __forceinline__ void tdm_load_b(const _Float16* gsrc, const _Float16* ldst) {
  unsigned lds = (unsigned)(unsigned long long)ldst;       // low 32 bits = LDS offset
  unsigned long long ga = (unsigned long long)gsrc;
  v4u g0 = { 1u,                                           // count=1, is_restore=0
             lds,                                          // lds_addr [63:32]
             (unsigned)ga,                                 // global_addr [95:64]
             (unsigned)((ga >> 32) & 0x1FFFFFFull) | 0x80000000u }; // ga[56:32] | type=2
  v8i_t g1 = { (int)0x00010000u,   // dw0: wg_mask=0, data_size=1 (2B)
               (int)0xFFFF0000u,   // dw1: abar=0, tensor_dim0[15:0]=0xFFFF
               (int)0xFFFF7FFFu,   // dw2: tensor_dim0[31:16]=0x7FFF, tensor_dim1[15:0]=0xFFFF
               (int)0x00207FFFu,   // dw3: tensor_dim1[31:16]=0x7FFF, tile_dim0=32
               (int)CHUNK,         // dw4: tile_dim1=128, tile_dim2=0
               (int)DD,            // dw5: tensor_dim0_stride[31:0]=1024
               0, 0 };             // dw6/7: stride hi, dim1_stride = 0
#if TDM_ARITY == 6
  __builtin_amdgcn_tensor_load_to_lds(g0, g1, (v4i_t)0, (v4i_t)0, (v8i_t)0, 0);
#else
  __builtin_amdgcn_tensor_load_to_lds(g0, g1, (v4i_t)0, (v4i_t)0, 0);
#endif
}
#endif

// ---------------- helpers ----------------
__device__ __forceinline__ bool beats(float v, int i, float v2, int i2) {
  return (v > v2) || (v == v2 && i < i2);   // val desc, idx asc (jax top_k ties)
}
__device__ __forceinline__ void push4(float* tv, int* ti, float v, int idx) {
  if (!beats(v, idx, tv[3], ti[3])) return;
  tv[3] = v; ti[3] = idx;
  for (int j = 3; j > 0 && beats(tv[j], ti[j], tv[j-1], ti[j-1]); --j) {
    float fv = tv[j]; tv[j] = tv[j-1]; tv[j-1] = fv;
    int   iv = ti[j]; ti[j] = ti[j-1]; ti[j-1] = iv;
  }
}

// ---------------- one-time: out_w f32 [D][V] -> f16 transposed [V][D] ----------------
__global__ __launch_bounds__(256) void wconvert_kernel(const float* __restrict__ out_w,
                                                       _Float16* __restrict__ wBt) {
  size_t t = (size_t)blockIdx.x * 256 + threadIdx.x;
  size_t e0 = t * 8;
  for (int u = 0; u < 8; ++u) {
    size_t e = e0 + u;                       // grid sized exactly: e < DD*VV
    int d = (int)(e / VV), v = (int)(e % VV);
    wBt[(size_t)v * DD + d] = (_Float16)out_w[e];
  }
}

// ---------------- one-time: context + state init ----------------
__global__ __launch_bounds__(256) void init_kernel(
    const int* __restrict__ src_inputs, const int* __restrict__ src_sizes,
    const int* __restrict__ first_tokens, const int* __restrict__ src_langs,
    const int* __restrict__ tgt_langs, const int* __restrict__ pad_idx_p,
    const float* __restrict__ emb, const float* __restrict__ lang_emb,
    float* __restrict__ cbase, int* tokens, float* scores, int* lastA,
    int* eosF, int* firstE, int* maxLens)
{
  const int b = blockIdx.x, tid = threadIdx.x;
  const int cnt = src_sizes[b];              // mask = first cnt positions
  const float inv = 1.0f / (float)cnt;
  const int sl = src_langs[b], tl = tgt_langs[b];
  for (int d = tid; d < DD; d += 256) {
    float acc = 0.f;
    for (int s = 0; s < cnt; ++s)
      acc += emb[(size_t)src_inputs[b*SS + s] * DD + d];
    cbase[b*DD + d] = acc * inv + lang_emb[sl*DD + d] + lang_emb[tl*DD + d];
  }
  const int pad = pad_idx_p[0];
  for (int j = tid; j < WW*ML; j += 256) tokens[b*WW*ML + j] = pad;
  __syncthreads();
  if (tid < WW) {
    int r = b*WW + tid;
    tokens[r*ML + 0] = first_tokens[b];
    scores[r] = (tid == 0) ? 0.0f : NEGF;    // step-1 single-beam trick
    lastA[r]  = first_tokens[b];
    eosF[r] = 0; firstE[r] = 0;
  }
  if (tid == 0) {
    int ml = (int)(1.1 * (double)cnt + 5.0);
    maxLens[b] = ml < 512 ? ml : 512;
  }
}

// ---------------- per step: h = tanh(emb[last] + cbase), f16 ----------------
__global__ __launch_bounds__(256) void build_h_kernel(
    const float* __restrict__ emb, const float* __restrict__ cbase,
    const int* __restrict__ lastA, _Float16* __restrict__ hbuf)
{
  const int r = blockIdx.x, tid = threadIdx.x;
  const int b = r >> 2;
  const size_t ebase = (size_t)lastA[r] * DD;
  for (int j = 0; j < 4; ++j) {
    int d = tid*4 + j;
    hbuf[r*DD + d] = (_Float16)tanhf(emb[ebase + d] + cbase[b*DD + d]);
  }
}

// ---------------- per step: GEMM + fused online-softmax partials + top4 ----------------
// block = 256 thr (8 waves); wave w owns cols [chunk*128 + w*16, +16), all 4 M-tiles.
// B tiles (8 KB) DMA'd by TDM into double-buffered LDS; A tile staged via async-LDS.
__global__ __launch_bounds__(256) void gemm_topk_kernel(
    const _Float16* __restrict__ hbuf, const _Float16* __restrict__ wBt,
    float* __restrict__ pm, float* __restrict__ ps,
    float* __restrict__ ptv, int* __restrict__ pti)
{
  __shared__ __align__(16) _Float16 sA[64*KTILE];      // 4 KB
#if USE_TDM
  __shared__ __align__(16) _Float16 sB[2][CHUNK*KTILE]; // 2 x 8 KB
#endif
  __shared__ float sLog[64*CHUNK];                     // 32 KB
  __shared__ float sRm[64*4], sRs[64*4];
  __shared__ float sRtv[64*4*4]; __shared__ int sRti[64*4*4];

  const int tid  = threadIdx.x;
  const int wave = tid >> 5, lane = tid & 31;
  const int hi   = lane >> 4;       // lane group: 0 -> lanes 0-15, 1 -> 16-31
  const int l15  = lane & 15;
  const int chunk = blockIdx.x;

  v8f acc[4] = {};                  // 4 M-tiles of 16x16 f32
  const int arow = tid >> 2, aseg = tid & 3;
  const _Float16* btile = wBt + (size_t)chunk*CHUNK*DD;  // 128-row block start

#if USE_TDM
  if (wave == 0) {                  // prologue: DMA first B tile, wait on TENSORcnt
    tdm_load_b(btile + 0, &sB[0][0]);
    wait_tensorcnt0();
  }
#endif

  for (int it = 0; it < NKIT; ++it) {
    const int k0 = it*KTILE;
    __syncthreads();   // prev sA reads done; sB[(it+1)&1] readers done; cur sB visible
#if USE_TDM
    if (wave == 0 && it + 1 < NKIT)            // overlap: DMA next B tile
      tdm_load_b(btile + (k0 + KTILE), &sB[(it+1)&1][0]);
#endif
    // --- stage A tile (64x32 halves): 256 threads x one 16B segment ---
    const _Float16* gA = hbuf + arow*DD + k0 + aseg*8;
    _Float16*       lA = sA + arow*KTILE + aseg*8;
#if USE_ASYNC_LDS
    {
      unsigned ldsa = (unsigned)(unsigned long long)lA;  // low 32 bits = LDS offset
      asm volatile("global_load_async_to_lds_b128 %0, %1, off"
                   :: "v"(ldsa), "v"(gA) : "memory");
      asm volatile("s_wait_asynccnt 0x0" ::: "memory");
    }
#else
    *(v8h*)lA = *(const v8h*)gA;
#endif
    __syncthreads();

    // --- B fragment (32x16): lane = col, K = hi*16 + 0..15 contiguous ---
#if USE_TDM
    const _Float16* bp = &sB[it & 1][(wave*16 + l15)*KTILE + hi*16];
#else
    const _Float16* bp = wBt + (size_t)(chunk*CHUNK + wave*16 + l15)*DD + k0 + hi*16;
#endif
    v8h blo = *(const v8h*)bp;
    v8h bhi = *(const v8h*)(bp + 8);
    v16h bfrag = __builtin_shufflevector(blo, bhi, 0,1,2,3,4,5,6,7,8,9,10,11,12,13,14,15);

#pragma unroll
    for (int m = 0; m < 4; ++m) {
      // A 16x32 f16 layout: lanes<16 K 0-15, lanes>=16 K 16-31 (contiguous, matches B)
      const _Float16* ap = sA + (m*16 + l15)*KTILE + hi*16;
      v8h alo = *(const v8h*)ap;
      v8h ahi = *(const v8h*)(ap + 8);
      v16h afrag = __builtin_shufflevector(alo, ahi, 0,1,2,3,4,5,6,7,8,9,10,11,12,13,14,15);
      acc[m] = __builtin_amdgcn_wmma_f32_16x16x32_f16(
          false, afrag, false, bfrag, 0, acc[m], false, false);
    }
#if USE_TDM
    if (wave == 0) wait_tensorcnt0();          // next tile landed before next barrier
#endif
  }

  // --- spill logits tile to LDS (D layout: VGPR g -> row g / g+8) ---
  __syncthreads();
#pragma unroll
  for (int m = 0; m < 4; ++m)
#pragma unroll
    for (int g = 0; g < 8; ++g) {
      int row = m*16 + g + hi*8;
      sLog[row*CHUNK + wave*16 + l15] = acc[m][g];
    }
  __syncthreads();

  // --- per-row chunk reduction: 4 threads/row, 32 cols each ---
  {
    const int row = tid >> 2, sub = tid & 3;
    float M = FNEG, S = 0.f;
    float tv[4] = {FNEG,FNEG,FNEG,FNEG};
    int   ti[4] = {0x7fffffff,0x7fffffff,0x7fffffff,0x7fffffff};
    const float* rowp = sLog + row*CHUNK + sub*32;
    for (int c = 0; c < 32; ++c) {
      float x = rowp[c];
      if (x > M) { S = S*__expf(M - x) + 1.f; M = x; }
      else       { S += __expf(x - M); }
      push4(tv, ti, x, chunk*CHUNK + sub*32 + c);
    }
    int slot = row*4 + sub;
    sRm[slot] = M; sRs[slot] = S;
    for (int j = 0; j < 4; ++j) { sRtv[slot*4+j] = tv[j]; sRti[slot*4+j] = ti[j]; }
  }
  __syncthreads();
  if ((tid & 3) == 0) {
    const int row = tid >> 2;
    float M = FNEG, S = 0.f;
    float tv[4] = {FNEG,FNEG,FNEG,FNEG};
    int   ti[4] = {0x7fffffff,0x7fffffff,0x7fffffff,0x7fffffff};
    for (int p = 0; p < 4; ++p) {
      int slot = row*4 + p;
      float m2 = sRm[slot], s2 = sRs[slot];
      if (m2 > M) { S = S*__expf(M - m2) + s2; M = m2; }
      else        { S += s2*__expf(m2 - M); }
      for (int j = 0; j < 4; ++j) push4(tv, ti, sRtv[slot*4+j], sRti[slot*4+j]);
    }
    int o = chunk*NROW + row;
    pm[o] = M; ps[o] = S;
    for (int j = 0; j < 4; ++j) { ptv[o*4+j] = tv[j]; pti[o*4+j] = ti[j]; }
  }
}

// ---------------- per step: merge partials, beam select, state update ----------------
__global__ __launch_bounds__(256) void update_kernel(
    const float* __restrict__ pm, const float* __restrict__ ps,
    const float* __restrict__ ptv, const int* __restrict__ pti,
    int* tokens, int* tokTmp, float* scores, int* lastA,
    int* eosF, int* firstE, const int* __restrict__ maxLens,
    const int* __restrict__ pad_idx_p, int step)
{
  __shared__ float cV[NROW][4]; __shared__ int cI[NROW][4];
  __shared__ int   sBeam[BB][WW]; __shared__ int sWord[BB][WW];
  __shared__ float sVal[BB][WW];
  __shared__ int   sAll;
  const int tid = threadIdx.x;
  const int pad = pad_idx_p[0];

  if (tid == 0) {
    int a = 1;
    for (int r = 0; r < NROW; ++r) a &= (eosF[r] != 0);
    sAll = a;                                  // emulate "all_done" freeze
  }
  if (tid < NROW) {
    const int r = tid;
    float M = FNEG, S = 0.f;
    float tv[4] = {FNEG,FNEG,FNEG,FNEG};
    int   ti[4] = {0x7fffffff,0x7fffffff,0x7fffffff,0x7fffffff};
    for (int c = 0; c < NCHUNK; ++c) {
      int o = c*NROW + r;
      float m2 = pm[o], s2 = ps[o];
      if (m2 > M) { S = S*__expf(M - m2) + s2; M = m2; }
      else        { S += s2*__expf(m2 - M); }
      for (int j = 0; j < 4; ++j) push4(tv, ti, ptv[o*4+j], pti[o*4+j]);
    }
    const int b = r >> 2, w = r & 3;
    const int eos = eosF[r];
    const int reached = (maxLens[b] < step + 1);
    const int zero = eos || (reached && step > 1);
    float len = eos ? (float)(firstE[r] + 6) : (float)(step + 5);
    float pen = powf(len / 6.0f, 0.8f);        // GNMT length penalty
    float sc = scores[r];
    if (zero) {                                // logp == 0 rows: ties at words 0..3
      for (int j = 0; j < 4; ++j) { cV[r][j] = sc / pen; cI[r][j] = w*VV + j; }
    } else {
      float lse = M + logf(S);
      for (int j = 0; j < 4; ++j) {
        cV[r][j] = (sc + (tv[j] - lse)) / pen;
        cI[r][j] = w*VV + ti[j];
      }
    }
  }
  __syncthreads();
  if (sAll) return;                            // frozen state (uniform)

  if (tid < BB) {
    const int b = tid;
    float tv[4] = {FNEG,FNEG,FNEG,FNEG};
    int   ti[4] = {0x7fffffff,0x7fffffff,0x7fffffff,0x7fffffff};
    for (int w = 0; w < WW; ++w)
      for (int j = 0; j < 4; ++j)
        push4(tv, ti, cV[b*WW + w][j], cI[b*WW + w][j]);
    const int force_pad = (maxLens[b] < step + 1) && (step > 1);
    for (int k = 0; k < WW; ++k) {
      int idx = force_pad ? pad : ti[k];       // reference overrides idx only
      sBeam[b][k] = idx / VV;
      sWord[b][k] = idx % VV;
      sVal[b][k]  = tv[k];
    }
  }
  __syncthreads();
  // gather-then-set into tokTmp (old tokens stay intact while gathering)
  for (int e = tid; e < BB*WW*ML; e += 256) {
    int p = e % ML, bw = e / ML;
    int b = bw / WW, w = bw % WW;
    int src = b*WW + sBeam[b][w];
    tokTmp[e] = (p == step) ? sWord[b][w] : tokens[src*ML + p];
  }
  __syncthreads();
  for (int e = tid; e < BB*WW*ML; e += 256) tokens[e] = tokTmp[e];
  if (tid < NROW) {
    int b = tid >> 2, w = tid & 3;
    scores[tid] = sVal[b][w];
    lastA[tid]  = sWord[b][w];
  }
  __syncthreads();
  if (tid < NROW) {                            // flags for NEXT step
    int fe = 0, hit = 0;
    const int* row = tokens + tid*ML;
    for (int p = 0; p <= step; ++p)
      if (row[p] == EOS_ID) { fe = p; hit = 1; break; }
    eosF[tid] = hit; firstE[tid] = fe;
  }
}

// ---------------- output: tokens[:,0,:] (int bits) then scores ----------------
__global__ __launch_bounds__(256) void finalize_kernel(
    const int* __restrict__ tokens, const float* __restrict__ scores,
    float* __restrict__ outF)
{
  const int tid = threadIdx.x;
  int* outI = (int*)outF;
  for (int e = tid; e < BB*ML; e += 256) {
    int b = e / ML, p = e % ML;
    outI[e] = tokens[(b*WW + 0)*ML + p];
  }
  if (tid < NROW) outF[BB*ML + tid] = scores[tid];
}

// ---------------- host ----------------
extern "C" void kernel_launch(void* const* d_in, const int* in_sizes, int n_in,
                              void* d_out, int out_size, void* d_ws, size_t ws_size,
                              hipStream_t stream)
{
  (void)in_sizes; (void)n_in; (void)out_size; (void)ws_size;
  const int*   src_inputs   = (const int*)d_in[0];
  const int*   src_sizes    = (const int*)d_in[1];
  const int*   first_tokens = (const int*)d_in[2];
  /* d_in[3] = src_mask: recomputed from src_sizes, unused */
  const int*   src_langs    = (const int*)d_in[4];
  const int*   tgt_langs    = (const int*)d_in[5];
  const int*   pad_idx      = (const int*)d_in[6];
  const float* emb          = (const float*)d_in[7];
  const float* lang_emb     = (const float*)d_in[8];
  const float* out_w        = (const float*)d_in[9];

  char* ws = (char*)d_ws;
  size_t off = 0;
  auto take = [&](size_t bytes) {
    size_t o = off; off += (bytes + 255) & ~(size_t)255; return o;
  };
  _Float16* wBt    = (_Float16*)(ws + take((size_t)VV*DD*2));      // ~62.5 MiB (L2-resident)
  float*    cbase  = (float*)   (ws + take((size_t)BB*DD*4));
  _Float16* hbuf   = (_Float16*)(ws + take((size_t)NROW*DD*2));
  float*    pm     = (float*)   (ws + take((size_t)NCHUNK*NROW*4));
  float*    psum   = (float*)   (ws + take((size_t)NCHUNK*NROW*4));
  float*    ptv    = (float*)   (ws + take((size_t)NCHUNK*NROW*4*4));
  int*      pti    = (int*)     (ws + take((size_t)NCHUNK*NROW*4*4));
  int*      tokens = (int*)     (ws + take((size_t)BB*WW*ML*4));
  int*      tokTmp = (int*)     (ws + take((size_t)BB*WW*ML*4));
  float*    scores = (float*)   (ws + take(NROW*4));
  int*      lastA  = (int*)     (ws + take(NROW*4));
  int*      eosF   = (int*)     (ws + take(NROW*4));
  int*      firstE = (int*)     (ws + take(NROW*4));
  int*      maxLens= (int*)     (ws + take(BB*4));

  wconvert_kernel<<<(VV*DD)/(256*8), 256, 0, stream>>>(out_w, wBt);
  init_kernel<<<BB, 256, 0, stream>>>(src_inputs, src_sizes, first_tokens,
      src_langs, tgt_langs, pad_idx, emb, lang_emb,
      cbase, tokens, scores, lastA, eosF, firstE, maxLens);

  for (int i = 1; i < ML; ++i) {
    build_h_kernel<<<NROW, 256, 0, stream>>>(emb, cbase, lastA, hbuf);
    gemm_topk_kernel<<<NCHUNK, 256, 0, stream>>>(hbuf, wBt, pm, psum, ptv, pti);
    update_kernel<<<1, 256, 0, stream>>>(pm, psum, ptv, pti, tokens, tokTmp,
        scores, lastA, eosF, firstE, maxLens, pad_idx, i);
  }
  finalize_kernel<<<1, 256, 0, stream>>>(tokens, scores, (float*)d_out);
}
